// MultiHeadSelfAttention_61529701483038
// MI455X (gfx1250) — compile-verified, matrix operand order checked by reading, not run on hardware
//
#include <hip/hip_runtime.h>
#include <hip/hip_bf16.h>

// MHSA forward for MI455X (gfx1250, wave32, WMMA + async-to-LDS staging).
// All GEMMs via v_wmma_f32_16x16x32_f16 (f16 operands, f32 accum).
// Block-shared operands staged in LDS with global_load_async_to_lds_b128
// (ASYNCcnt), double-buffered; per-wave A fragments register-pipelined.

typedef _Float16 f16;
typedef _Float16 v16h __attribute__((ext_vector_type(16)));
typedef _Float16 v8h  __attribute__((ext_vector_type(8)));
typedef float    v8f  __attribute__((ext_vector_type(8)));

#define DIMC   768
#define NSEQ   1024
#define BATCH  16
#define HEADS  12
#define HD     64
#define QKVC   (3*DIMC)
#define NROWS  (BATCH*NSEQ)      // 16384

#define WMMA_F16(a,b,c) \
  __builtin_amdgcn_wmma_f32_16x16x32_f16(false,(a),false,(b),(short)0,(c),false,false)

// ---- CDNA5 async copy: 32 bytes per lane, global -> LDS, ASYNCcnt tracked ----
__device__ __forceinline__ void async_cp32(const f16* gsrc, f16* ldst) {
  uint64_t g = (uint64_t)(uintptr_t)gsrc;
  uint32_t l = (uint32_t)(uintptr_t)(void*)ldst;   // low 32 bits = LDS offset
  asm volatile(
      "global_load_async_to_lds_b128 %0, %1, off\n\t"
      "global_load_async_to_lds_b128 %0, %1, off offset:16"
      :: "v"(l), "v"(g) : "memory");
}
__device__ __forceinline__ void async_wait_all() {
  asm volatile("s_wait_asynccnt 0" ::: "memory");
}

// Load a 16x32 f16 fragment (A-operand layout; B operands are passed as B^T in
// the same layout) from a row-major matrix with leading dim ld (multiple of 8).
// ISA layout: lane = M%16; lanes 0-15 hold K 0..7 & 16..23, lanes 16-31 hold
// K 8..15 & 24..31  -> two contiguous 16-byte loads per lane.
__device__ __forceinline__ v16h load_frag(const f16* tile, int ld) {
  const int lane = threadIdx.x & 31;
  const int r = lane & 15;
  const int hi = lane >> 4;
  const f16* p = tile + (size_t)r * ld + hi * 8;
  v8h a = *(const v8h*)p;
  v8h b = *(const v8h*)(p + 16);
  v16h out;
#pragma unroll
  for (int i = 0; i < 8; ++i) { out[i] = a[i]; out[i + 8] = b[i]; }
  return out;
}

// ---------------- conversion kernels ----------------
__global__ void k_cvt_f16(const float* __restrict__ s, f16* __restrict__ d, int n) {
  int i = blockIdx.x * 256 + threadIdx.x;
  if (i < n) d[i] = (f16)s[i];
}

// dst[c][r] = (f16) src[r][c]   (src is R x C row-major)
__global__ void k_cvt_t_f16(const float* __restrict__ s, f16* __restrict__ d, int R, int C) {
  int i = blockIdx.x * 256 + threadIdx.x;
  if (i < R * C) {
    int r = i / C, c = i - r * C;
    d[(size_t)c * R + r] = (f16)s[i];
  }
}

// ---------------- QKV projection ----------------
// qkv = xh[16384x768] @ wT^T + bias, scattered into Q, K ([bh][n][d]) and
// Vt ([bh][d][n]).  Block: 4 waves, 64x64 tile; B staged in LDS (async,
// double-buffered, shared by all 4 waves). grid (36, 256).
__global__ __launch_bounds__(128) void k_qkv_gemm(
    const f16* __restrict__ xh, const f16* __restrict__ wT,
    const float* __restrict__ bias,
    f16* __restrict__ Q, f16* __restrict__ K, f16* __restrict__ Vt) {
  __shared__ f16 ldsB[2][64][32];               // 8 KB, double buffer
  const int t = threadIdx.x;
  const int wv = t >> 5;
  const int lane = t & 31;
  const int row0 = blockIdx.y * 64 + wv * 16;
  const int col0 = blockIdx.x * 64;

  // copy assignment: 64 cols x 2 threads, 32B each
  const int ccol = t >> 1, chalf = t & 1;
  const f16* wsrc = wT + (size_t)(col0 + ccol) * DIMC + chalf * 16;
  f16* wdst[2] = {&ldsB[0][ccol][chalf * 16], &ldsB[1][ccol][chalf * 16]};

  async_cp32(wsrc, wdst[0]);                    // prefetch k=0 tile
  const f16* arow = xh + (size_t)row0 * DIMC;
  v16h a_cur = load_frag(arow, DIMC);
  async_wait_all();
  __syncthreads();

  v8f acc0 = {}, acc1 = {}, acc2 = {}, acc3 = {};
#pragma unroll 1
  for (int k = 0; k < DIMC; k += 32) {
    const int cur = (k >> 5) & 1;
    v16h a_nxt = a_cur;
    if (k + 32 < DIMC) {
      async_cp32(wsrc + (k + 32), wdst[cur ^ 1]);     // fill other buffer
      a_nxt = load_frag(arow + k + 32, DIMC);         // prefetch next A frag
    }
    v16h b0 = load_frag(&ldsB[cur][ 0][0], 32);
    v16h b1 = load_frag(&ldsB[cur][16][0], 32);
    v16h b2 = load_frag(&ldsB[cur][32][0], 32);
    v16h b3 = load_frag(&ldsB[cur][48][0], 32);
    acc0 = WMMA_F16(a_cur, b0, acc0);
    acc1 = WMMA_F16(a_cur, b1, acc1);
    acc2 = WMMA_F16(a_cur, b2, acc2);
    acc3 = WMMA_F16(a_cur, b3, acc3);
    a_cur = a_nxt;
    if (k + 32 < DIMC) { async_wait_all(); __syncthreads(); }
  }

  const int cl = lane & 15;
  const int hi = lane >> 4;
  v8f accs[4] = {acc0, acc1, acc2, acc3};
#pragma unroll
  for (int c = 0; c < 4; ++c) {
    const int col = col0 + c * 16 + cl;
    const float bv = bias[col];
    const int part = col / DIMC;          // 0=Q 1=K 2=V
    const int cc = col - part * DIMC;
    const int h = cc / HD, d = cc - h * HD;
#pragma unroll
    for (int v = 0; v < 8; ++v) {
      const int row = row0 + v + 8 * hi;  // global token
      const int b = row >> 10, n = row & (NSEQ - 1);
      const f16 hv = (f16)(accs[c][v] + bv);
      const size_t bh = (size_t)(b * HEADS + h);
      if (part == 0)      Q[(bh * NSEQ + n) * HD + d] = hv;
      else if (part == 1) K[(bh * NSEQ + n) * HD + d] = hv;
      else                Vt[(bh * HD + d) * NSEQ + n] = hv;
    }
  }
}

// ---------------- flash attention ----------------
// grid (192 = b*h, 16 = q-block of 64 rows); 4 waves, each owns 16 q rows.
// K (32x64) and Vt (64x32) tiles staged in LDS by async copies, double-buffered.
__global__ __launch_bounds__(128) void k_attn(
    const f16* __restrict__ Q, const f16* __restrict__ K,
    const f16* __restrict__ Vt, f16* __restrict__ AO) {
  __shared__ f16 ldsK[2][32][64];   // 8 KB
  __shared__ f16 ldsV[2][64][32];   // 8 KB
  __shared__ f16 ldsP[4][16][40];   // per-wave P transpose tile, ld=40 (16B align)
  const int t = threadIdx.x;
  const int wv = t >> 5;
  const int lane = t & 31;
  const int cl = lane & 15;
  const int hi = lane >> 4;
  const int bh = blockIdx.x;
  const int row0 = blockIdx.y * 64 + wv * 16;

  const f16* Qbh = Q  + (size_t)bh * NSEQ * HD;
  const f16* Kbh = K  + (size_t)bh * NSEQ * HD;
  const f16* Vbh = Vt + (size_t)bh * HD * NSEQ;

  // copy assignments: K tile 32 rows x 4 chunks(32B); V tile 64 rows x 2 chunks
  const int krow = t >> 2, kq = t & 3;
  const int vrow = t >> 1, vh = t & 1;
  const f16* ksrc = Kbh + (size_t)krow * HD + kq * 16;
  const f16* vsrc = Vbh + (size_t)vrow * NSEQ + vh * 16;
  f16* kdst[2] = {&ldsK[0][krow][kq * 16], &ldsK[1][krow][kq * 16]};
  f16* vdst[2] = {&ldsV[0][vrow][vh * 16], &ldsV[1][vrow][vh * 16]};

  async_cp32(ksrc, kdst[0]);                 // prefetch j=0 tiles
  async_cp32(vsrc, vdst[0]);

  const v16h qa0 = load_frag(Qbh + (size_t)row0 * HD +  0, HD);
  const v16h qa1 = load_frag(Qbh + (size_t)row0 * HD + 32, HD);

  v8f O0 = {}, O1 = {}, O2 = {}, O3 = {};
  float m[8], l[8];
#pragma unroll
  for (int v = 0; v < 8; ++v) { m[v] = -3.0e38f; l[v] = 0.0f; }
  const float scale = 0.125f;   // 1/sqrt(64)

  async_wait_all();
  __syncthreads();

#pragma unroll 1
  for (int j = 0; j < NSEQ; j += 32) {
    const int cur = (j >> 5) & 1;
    if (j + 32 < NSEQ) {                     // fill other buffer while computing
      async_cp32(ksrc + (size_t)(j + 32) * HD, kdst[cur ^ 1]);
      async_cp32(vsrc + (j + 32), vdst[cur ^ 1]);
    }

    // S = Q . K^T (scaled): 2 column tiles of 16, head-dim 64 in two chunks
    v16h b00 = load_frag(&ldsK[cur][ 0][ 0], 64);
    v16h b01 = load_frag(&ldsK[cur][ 0][32], 64);
    v16h b10 = load_frag(&ldsK[cur][16][ 0], 64);
    v16h b11 = load_frag(&ldsK[cur][16][32], 64);
    v8f s0 = {}, s1 = {};
    s0 = WMMA_F16(qa0, b00, s0);
    s0 = WMMA_F16(qa1, b01, s0);
    s1 = WMMA_F16(qa0, b10, s1);
    s1 = WMMA_F16(qa1, b11, s1);

    // online softmax: C layout puts columns in lanes -> row reductions are
    // xor-shuffles over the 16-lane half (rows v+8*hi live per VGPR index v)
    float f_[8], p0[8], p1[8];
#pragma unroll
    for (int v = 0; v < 8; ++v) {
      const float a = s0[v] * scale;
      const float b = s1[v] * scale;
      float tmx = fmaxf(a, b);
      tmx = fmaxf(tmx, __shfl_xor(tmx, 1, 32));
      tmx = fmaxf(tmx, __shfl_xor(tmx, 2, 32));
      tmx = fmaxf(tmx, __shfl_xor(tmx, 4, 32));
      tmx = fmaxf(tmx, __shfl_xor(tmx, 8, 32));
      const float mn = fmaxf(m[v], tmx);
      f_[v] = __expf(m[v] - mn);
      m[v] = mn;
      p0[v] = __expf(a - mn);
      p1[v] = __expf(b - mn);
      float s = p0[v] + p1[v];
      s += __shfl_xor(s, 1, 32);
      s += __shfl_xor(s, 2, 32);
      s += __shfl_xor(s, 4, 32);
      s += __shfl_xor(s, 8, 32);
      l[v] = l[v] * f_[v] + s;
    }
#pragma unroll
    for (int v = 0; v < 8; ++v) {
      O0[v] *= f_[v]; O1[v] *= f_[v]; O2[v] *= f_[v]; O3[v] *= f_[v];
    }

    // transpose P (C layout -> A layout) through per-wave LDS tile
#pragma unroll
    for (int v = 0; v < 8; ++v) {
      ldsP[wv][v + 8 * hi][cl]      = (f16)p0[v];
      ldsP[wv][v + 8 * hi][16 + cl] = (f16)p1[v];
    }
    asm volatile("" ::: "memory");           // keep DS store->load order
    const v16h pa = load_frag(&ldsP[wv][0][0], 40);
    asm volatile("" ::: "memory");

    // O += P . V  (B fragments from staged Vt rows: contiguous)
    v16h vb0 = load_frag(&ldsV[cur][ 0][0], 32);
    v16h vb1 = load_frag(&ldsV[cur][16][0], 32);
    v16h vb2 = load_frag(&ldsV[cur][32][0], 32);
    v16h vb3 = load_frag(&ldsV[cur][48][0], 32);
    O0 = WMMA_F16(pa, vb0, O0);
    O1 = WMMA_F16(pa, vb1, O1);
    O2 = WMMA_F16(pa, vb2, O2);
    O3 = WMMA_F16(pa, vb3, O3);

    if (j + 32 < NSEQ) { async_wait_all(); __syncthreads(); }
  }

  // normalize and scatter to attn-out [b*n][768] (concat heads) in f16
  const int h = bh % HEADS;
  const int b = bh / HEADS;
  v8f Os[4] = {O0, O1, O2, O3};
#pragma unroll
  for (int c = 0; c < 4; ++c) {
#pragma unroll
    for (int v = 0; v < 8; ++v) {
      const int n = row0 + v + 8 * hi;
      const float val = Os[c][v] / l[v];
      AO[((size_t)(b * NSEQ + n)) * DIMC + h * HD + c * 16 + cl] = (f16)val;
    }
  }
}

// ---------------- output projection ----------------
// out = AO[16384x768] @ wprojT^T + bias  (f32 result). grid (12, 256).
__global__ __launch_bounds__(128) void k_proj_gemm(
    const f16* __restrict__ A, const f16* __restrict__ wT,
    const float* __restrict__ bias, float* __restrict__ out) {
  __shared__ f16 ldsB[2][64][32];
  const int t = threadIdx.x;
  const int wv = t >> 5;
  const int lane = t & 31;
  const int row0 = blockIdx.y * 64 + wv * 16;
  const int col0 = blockIdx.x * 64;

  const int ccol = t >> 1, chalf = t & 1;
  const f16* wsrc = wT + (size_t)(col0 + ccol) * DIMC + chalf * 16;
  f16* wdst[2] = {&ldsB[0][ccol][chalf * 16], &ldsB[1][ccol][chalf * 16]};

  async_cp32(wsrc, wdst[0]);
  const f16* arow = A + (size_t)row0 * DIMC;
  v16h a_cur = load_frag(arow, DIMC);
  async_wait_all();
  __syncthreads();

  v8f acc0 = {}, acc1 = {}, acc2 = {}, acc3 = {};
#pragma unroll 1
  for (int k = 0; k < DIMC; k += 32) {
    const int cur = (k >> 5) & 1;
    v16h a_nxt = a_cur;
    if (k + 32 < DIMC) {
      async_cp32(wsrc + (k + 32), wdst[cur ^ 1]);
      a_nxt = load_frag(arow + k + 32, DIMC);
    }
    v16h b0 = load_frag(&ldsB[cur][ 0][0], 32);
    v16h b1 = load_frag(&ldsB[cur][16][0], 32);
    v16h b2 = load_frag(&ldsB[cur][32][0], 32);
    v16h b3 = load_frag(&ldsB[cur][48][0], 32);
    acc0 = WMMA_F16(a_cur, b0, acc0);
    acc1 = WMMA_F16(a_cur, b1, acc1);
    acc2 = WMMA_F16(a_cur, b2, acc2);
    acc3 = WMMA_F16(a_cur, b3, acc3);
    a_cur = a_nxt;
    if (k + 32 < DIMC) { async_wait_all(); __syncthreads(); }
  }

  const int cl = lane & 15;
  const int hi = lane >> 4;
  v8f accs[4] = {acc0, acc1, acc2, acc3};
#pragma unroll
  for (int c = 0; c < 4; ++c) {
    const int col = col0 + c * 16 + cl;
    const float bv = bias[col];
#pragma unroll
    for (int v = 0; v < 8; ++v) {
      const int row = row0 + v + 8 * hi;
      out[(size_t)row * DIMC + col] = accs[c][v] + bv;
    }
  }
}

// ---------------- host launch ----------------
extern "C" void kernel_launch(void* const* d_in, const int* in_sizes, int n_in,
                              void* d_out, int out_size, void* d_ws, size_t ws_size,
                              hipStream_t stream) {
  const float* x      = (const float*)d_in[0];   // [16,1024,768]
  const float* w_qkv  = (const float*)d_in[1];   // [768,2304]
  const float* b_qkv  = (const float*)d_in[2];   // [2304]
  const float* w_proj = (const float*)d_in[3];   // [768,768]
  const float* b_proj = (const float*)d_in[4];   // [768]
  float* out = (float*)d_out;

  f16* ws = (f16*)d_ws;
  size_t o = 0;
  f16* xh     = ws + o; o += (size_t)NROWS * DIMC;       // x in f16
  f16* wqkvT  = ws + o; o += (size_t)QKVC * DIMC;        // w_qkv^T f16
  f16* wprojT = ws + o; o += (size_t)DIMC * DIMC;        // w_proj^T f16
  f16* Qb     = ws + o; o += (size_t)BATCH * HEADS * NSEQ * HD;
  f16* Kb     = ws + o; o += (size_t)BATCH * HEADS * NSEQ * HD;
  f16* Vt     = ws + o; o += (size_t)BATCH * HEADS * HD * NSEQ;
  f16* AO     = ws + o; o += (size_t)NROWS * DIMC;       // attn output f16

  {
    int n = NROWS * DIMC;
    k_cvt_f16<<<(n + 255) / 256, 256, 0, stream>>>(x, xh, n);
  }
  {
    int n = DIMC * QKVC;
    k_cvt_t_f16<<<(n + 255) / 256, 256, 0, stream>>>(w_qkv, wqkvT, DIMC, QKVC);
  }
  {
    int n = DIMC * DIMC;
    k_cvt_t_f16<<<(n + 255) / 256, 256, 0, stream>>>(w_proj, wprojT, DIMC, DIMC);
  }

  k_qkv_gemm<<<dim3(QKVC / 64, NROWS / 64), 128, 0, stream>>>(
      xh, wqkvT, b_qkv, Qb, Kb, Vt);

  k_attn<<<dim3(BATCH * HEADS, NSEQ / 64), 128, 0, stream>>>(Qb, Kb, Vt, AO);

  k_proj_gemm<<<dim3(DIMC / 64, NROWS / 64), 128, 0, stream>>>(
      AO, wprojT, b_proj, out);
}
